// OlmoLMHeadCE_30253749633455
// MI455X (gfx1250) — matrix operand order repaired and checked by reading, not compiled, not run on hardware
//
#include <hip/hip_runtime.h>

// Fused LM-head cross-entropy for MI455X (gfx1250, wave32).
// logits = X[N,D] * W[D,V] via bf16 hi/lo-split WMMA (fp32-accurate, 3 products),
// streamed through a per-column-block partial softmax; logits never hit HBM.

#define IGNORE_INDEX (-100)

typedef __attribute__((ext_vector_type(16))) __bf16 v16bf;
typedef __attribute__((ext_vector_type(8)))  __bf16 v8bf;
typedef __attribute__((ext_vector_type(8)))  float  v8f;

#define BM 128
#define BN 128
#define KC 32
#define KP 40              // padded LDS row stride in elements: 80B rows -> 16B-aligned, conflict-free b128 reads
#define THREADS 256        // 8 wave32 waves
#define NEG_INF (-3.402823466e38f)

__device__ __forceinline__ v16bf join16(v8bf lo, v8bf hi) {
  return __builtin_shufflevector(lo, hi, 0,1,2,3,4,5,6,7,8,9,10,11,12,13,14,15);
}

__global__ __launch_bounds__(THREADS)
void ce_gemm_partial(const float* __restrict__ A, const float* __restrict__ W,
                     const int* __restrict__ labels,
                     float* __restrict__ pm, float* __restrict__ ps,
                     float* __restrict__ tgt,
                     int N, int D, int V, int CB)
{
  __shared__ __bf16 Ahi[BM][KP];
  __shared__ __bf16 Alo[BM][KP];
  __shared__ __bf16 Bhi[BN][KP];   // stored transposed: [n][k], K contiguous
  __shared__ __bf16 Blo[BN][KP];
  __shared__ int    sLabel[BM];
  __shared__ float  redm[BM][2];
  __shared__ float  reds[BM][2];

  const int tid    = threadIdx.x;
  const int lane   = tid & 31;
  const int wid    = tid >> 5;
  const int wave_m = wid & 3;      // 4 row strips of 32
  const int wave_n = wid >> 2;     // 2 col strips of 64

  const int row0 = blockIdx.x * BM;
  const int col0 = blockIdx.y * BN;

  if (tid < BM) {
    const int r = row0 + tid;
    sLabel[tid] = (r < N) ? labels[r] : IGNORE_INDEX;
  }

  v8f acc[2][4];
  const v8f vzero = {0.f,0.f,0.f,0.f,0.f,0.f,0.f,0.f};
#pragma unroll
  for (int i = 0; i < 2; ++i)
#pragma unroll
    for (int j = 0; j < 4; ++j) acc[i][j] = vzero;

  // Per-lane fragment geometry (ISA 16-bit A 16x32 / B 32x16 layouts)
  const int arow[2] = { wave_m*32 + 0*16 + (lane & 15),
                        wave_m*32 + 1*16 + (lane & 15) };
  const int khalfA  = (lane < 16) ? 0 : 8;    // A: lanes 0-15 hold K 0-7/16-23, lanes 16-31 hold K 8-15/24-31
  const int koffB   = (lane < 16) ? 0 : 8;    // B: lanes 0-15 hold K 0-7/16-23, lanes 16-31 hold K 8-15/24-31
  const int bcolBase = wave_n * 64;

  for (int k0 = 0; k0 < D; k0 += KC) {        // D assumed multiple of KC (2048/32)
    __syncthreads();
    // ---- stage A tile 128x32 fp32 -> bf16 hi/lo in LDS ----
    {
      const int r = tid >> 3;                 // 0..31
      const int c = (tid & 7) << 2;           // 0..28
#pragma unroll
      for (int p = 0; p < 4; ++p) {
        const int row = r + p*32;
        float4 x = make_float4(0.f,0.f,0.f,0.f);
        if (row0 + row < N)
          x = *reinterpret_cast<const float4*>(&A[(size_t)(row0 + row)*D + (k0 + c)]);
        const float xs[4] = {x.x, x.y, x.z, x.w};
#pragma unroll
        for (int q = 0; q < 4; ++q) {
          const __bf16 h = (__bf16)xs[q];
          Ahi[row][c+q] = h;
          Alo[row][c+q] = (__bf16)(xs[q] - (float)h);
        }
      }
    }
    // ---- stage W tile 32x128 fp32 -> transposed bf16 hi/lo in LDS ----
    {
      const int kk = tid >> 5;                // 0..7
      const int c  = (tid & 31) << 2;         // 0..124
#pragma unroll
      for (int p = 0; p < 4; ++p) {
        const int k = kk + p*8;
        float4 x = make_float4(0.f,0.f,0.f,0.f);
        if (col0 + c + 3 < V)                 // V multiple of 4; masked cols handled in epilogue
          x = *reinterpret_cast<const float4*>(&W[(size_t)(k0 + k)*V + (col0 + c)]);
        const float xs[4] = {x.x, x.y, x.z, x.w};
#pragma unroll
        for (int q = 0; q < 4; ++q) {
          const __bf16 h = (__bf16)xs[q];
          Bhi[c+q][k] = h;
          Blo[c+q][k] = (__bf16)(xs[q] - (float)h);
        }
      }
    }
    __syncthreads();

    // ---- fragments + WMMA ----
    v16bf ah[2], al[2];
#pragma unroll
    for (int tm = 0; tm < 2; ++tm) {
      const int r = arow[tm];
      ah[tm] = join16(*(const v8bf*)&Ahi[r][khalfA], *(const v8bf*)&Ahi[r][16 + khalfA]);
      al[tm] = join16(*(const v8bf*)&Alo[r][khalfA], *(const v8bf*)&Alo[r][16 + khalfA]);
    }
#pragma unroll
    for (int tn = 0; tn < 4; ++tn) {
      const int n = bcolBase + tn*16 + (lane & 15);
      const v16bf bh = join16(*(const v8bf*)&Bhi[n][koffB], *(const v8bf*)&Bhi[n][koffB + 16]);
      const v16bf bl = join16(*(const v8bf*)&Blo[n][koffB], *(const v8bf*)&Blo[n][koffB + 16]);
#pragma unroll
      for (int tm = 0; tm < 2; ++tm) {
        acc[tm][tn] = __builtin_amdgcn_wmma_f32_16x16x32_bf16(false, ah[tm], false, bh, (short)0, acc[tm][tn], false, false);
        acc[tm][tn] = __builtin_amdgcn_wmma_f32_16x16x32_bf16(false, ah[tm], false, bl, (short)0, acc[tm][tn], false, false);
        acc[tm][tn] = __builtin_amdgcn_wmma_f32_16x16x32_bf16(false, al[tm], false, bh, (short)0, acc[tm][tn], false, false);
      }
    }
  }

  // ---- epilogue: per-row (max, sum-exp) over this 128-col strip + target-logit gather ----
#pragma unroll
  for (int tm = 0; tm < 2; ++tm) {
#pragma unroll
    for (int g = 0; g < 8; ++g) {
      const int rowLocal = wave_m*32 + tm*16 + g + ((lane < 16) ? 0 : 8);
      const int lab = sLabel[rowLocal];
      float v[4];
      float m = NEG_INF;
#pragma unroll
      for (int tn = 0; tn < 4; ++tn) {
        const int col = col0 + wave_n*64 + tn*16 + (lane & 15);
        const float val = acc[tm][tn][g];
        v[tn] = (col < V) ? val : NEG_INF;
        m = fmaxf(m, v[tn]);
        if (col == lab && row0 + rowLocal < N) tgt[row0 + rowLocal] = val;  // unique writer
      }
#pragma unroll
      for (int off = 8; off > 0; off >>= 1) m = fmaxf(m, __shfl_xor(m, off, 32));
      float s = 0.f;
#pragma unroll
      for (int tn = 0; tn < 4; ++tn) s += expf(v[tn] - m);
#pragma unroll
      for (int off = 8; off > 0; off >>= 1) s += __shfl_xor(s, off, 32);
      if ((lane & 15) == 0) { redm[rowLocal][wave_n] = m; reds[rowLocal][wave_n] = s; }
    }
  }
  __syncthreads();
  if (tid < BM && row0 + tid < N) {
    const float m0 = redm[tid][0], m1 = redm[tid][1];
    const float s0 = reds[tid][0], s1 = reds[tid][1];
    const float m  = fmaxf(m0, m1);
    const float s  = s0*expf(m0 - m) + s1*expf(m1 - m);
    const size_t idx = (size_t)(row0 + tid) * CB + blockIdx.y;
    pm[idx] = m;
    ps[idx] = s;
  }
}

__global__ void ce_finalize(const float* __restrict__ pm, const float* __restrict__ ps,
                            const float* __restrict__ tgt, const int* __restrict__ labels,
                            float* __restrict__ out, float* __restrict__ lossr,
                            float* __restrict__ zr, int N, int CB)
{
  const int r = blockIdx.x * blockDim.x + threadIdx.x;
  if (r >= N) return;
  const float* pmr = pm + (size_t)r * CB;
  const float* psr = ps + (size_t)r * CB;
  float m = NEG_INF, s = 0.f;
  for (int i = 0; i < CB; ++i) {
    const float mi = pmr[i], si = psr[i];
    const float mn = fmaxf(m, mi);
    s = s * expf(m - mn) + si * expf(mi - mn);
    m = mn;
  }
  const float lse = m + logf(s);
  out[2 + r] = lse;                                 // lse output (per-row, regardless of mask)
  const bool valid = (labels[r] != IGNORE_INDEX);
  lossr[r] = valid ? (lse - tgt[r]) : 0.f;
  zr[r]    = valid ? (lse * lse)    : 0.f;
}

__global__ __launch_bounds__(1024)
void ce_reduce(const float* __restrict__ lossr, const float* __restrict__ zr,
               float* __restrict__ out, int N)
{
  __shared__ float sl[1024];
  __shared__ float sz[1024];
  const int t = threadIdx.x;
  float a = 0.f, b = 0.f;
  for (int i = t; i < N; i += 1024) { a += lossr[i]; b += zr[i]; }
  sl[t] = a; sz[t] = b;
  __syncthreads();
  for (int off = 512; off > 0; off >>= 1) {
    if (t < off) { sl[t] += sl[t + off]; sz[t] += sz[t + off]; }
    __syncthreads();
  }
  if (t == 0) { out[0] = sl[0]; out[1] = sz[0]; }   // deterministic fixed-tree sums
}

extern "C" void kernel_launch(void* const* d_in, const int* in_sizes, int n_in,
                              void* d_out, int out_size, void* d_ws, size_t ws_size,
                              hipStream_t stream) {
  const float* A      = (const float*)d_in[0];   // [N, D] fp32
  const float* W      = (const float*)d_in[1];   // [D, V] fp32
  const int*   labels = (const int*)d_in[2];     // [N]

  const int N = in_sizes[2];
  const int D = in_sizes[0] / N;
  const int V = in_sizes[1] / D;
  const int CB = (V + BN - 1) / BN;              // column blocks (393 for V=50304)

  float* out = (float*)d_out;                    // [loss, z_squared, lse[0..N)]

  float* pm    = (float*)d_ws;                   // [N, CB] partial row max
  float* ps    = pm  + (size_t)N * CB;           // [N, CB] partial sum-exp
  float* tgt   = ps  + (size_t)N * CB;           // [N] target logits
  float* lossr = tgt + N;                        // [N] per-row loss terms
  float* zr    = lossr + N;                      // [N] per-row lse^2 terms

  dim3 grid(N / BM, CB);                         // x-fastest: row-blocks of one W strip run adjacently -> W strip stays in L2
  ce_gemm_partial<<<grid, THREADS, 0, stream>>>(A, W, labels, pm, ps, tgt, N, D, V, CB);
  ce_finalize<<<(N + 255) / 256, 256, 0, stream>>>(pm, ps, tgt, labels, out, lossr, zr, N, CB);
  ce_reduce<<<1, 1024, 0, stream>>>(lossr, zr, out, N);
}
